// Attention_3427383902485
// MI455X (gfx1250) — compile-verified, matrix operand order checked by reading, not run on hardware
//
#include <hip/hip_runtime.h>

// ---------------------------------------------------------------------------
// MI455X (gfx1250) attention forward: bf16 WMMA everywhere, f32 accumulate.
// x:(B,L,HID) f32; Wq/Wk/Wv/Wo:(HID,HID) f32; cos/sin:(L,D) f32 -> out f32.
// ---------------------------------------------------------------------------

typedef __bf16 bf16_t;
typedef __attribute__((ext_vector_type(16))) __bf16 v16bf;
typedef __attribute__((ext_vector_type(8)))  float  v8f;
typedef __attribute__((ext_vector_type(4)))  unsigned int u32x4;

static constexpr int Bc   = 2;
static constexpr int Lc   = 2048;
static constexpr int Hc   = 16;
static constexpr int Dc   = 64;
static constexpr int HIDc = Hc * Dc;      // 1024
static constexpr int Mc   = Bc * Lc;      // 4096 rows in all GEMMs

union Frag { v16bf v; u32x4 q[2]; };

// A/B fragment loader for V_WMMA_F32_16X16X32_BF16.
// 16-bit A 16x32 layout: lanes 0-15 hold K={0..7,16..23}, lanes 16-31 hold
// K={8..15,24..31}; K pairs pack per VGPR -> per lane the fragment is two
// contiguous 16-byte spans of the row => two b128 loads.
__device__ __forceinline__ Frag load_frag(const bf16_t* __restrict__ base,
                                          int stride, int row, int k, int lane) {
    Frag f;
    const int koff = (lane & 16) ? 8 : 0;
    const bf16_t* p = base + (size_t)row * stride + k + koff;
    f.q[0] = *reinterpret_cast<const u32x4*>(p);
    f.q[1] = *reinterpret_cast<const u32x4*>(p + 16);
    return f;
}

__device__ __forceinline__ v8f wmma_bf16(Frag a, Frag b, v8f c) {
    return __builtin_amdgcn_wmma_f32_16x16x32_bf16(
        /*neg_a=*/false, a.v, /*neg_b=*/false, b.v,
        /*c_mod=*/(short)0, c, /*reuse_a=*/false, /*reuse_b=*/false);
}

// ---------------------------------------------------------------------------
// f32 -> bf16 convert
// ---------------------------------------------------------------------------
__global__ void cvt_f32_bf16(const float* __restrict__ src,
                             bf16_t* __restrict__ dst, int n) {
    int i = blockIdx.x * blockDim.x + threadIdx.x;
    if (i < n) dst[i] = (bf16_t)src[i];
}

// ---------------------------------------------------------------------------
// NT GEMM: C[M,N] f32 = A[M,K] bf16 * B[N,K]^T bf16   (M=4096, N=K=1024)
// block = 128 threads = 4 waves arranged 2x2; wave tile 32x64 (8 WMMA accums)
// grid = (N/128, M/64)
// ---------------------------------------------------------------------------
__global__ __launch_bounds__(128)
void gemm_nt_wmma(const bf16_t* __restrict__ A, const bf16_t* __restrict__ Bw,
                  float* __restrict__ C) {
    constexpr int K = HIDc, N = HIDc;
    const int lane = threadIdx.x & 31;
    const int wave = threadIdx.x >> 5;
    const int m0 = blockIdx.y * 64  + (wave >> 1) * 32;
    const int n0 = blockIdx.x * 128 + (wave & 1) * 64;

    const v8f vz = {0.f,0.f,0.f,0.f,0.f,0.f,0.f,0.f};
    v8f acc[2][4];
#pragma unroll
    for (int i = 0; i < 2; ++i)
#pragma unroll
        for (int j = 0; j < 4; ++j) acc[i][j] = vz;

    for (int kk = 0; kk < K; kk += 32) {
        Frag a[2], b[4];
#pragma unroll
        for (int i = 0; i < 2; ++i)
            a[i] = load_frag(A, K, m0 + i * 16 + (lane & 15), kk, lane);
#pragma unroll
        for (int j = 0; j < 4; ++j)
            b[j] = load_frag(Bw, K, n0 + j * 16 + (lane & 15), kk, lane);
#pragma unroll
        for (int i = 0; i < 2; ++i)
#pragma unroll
            for (int j = 0; j < 4; ++j)
                acc[i][j] = wmma_bf16(a[i], b[j], acc[i][j]);
    }

    // C layout: VGPR r -> row (lane<16 ? r : r+8); lane&15 -> col
#pragma unroll
    for (int i = 0; i < 2; ++i)
#pragma unroll
        for (int j = 0; j < 4; ++j)
#pragma unroll
            for (int r = 0; r < 8; ++r) {
                int row = m0 + i * 16 + ((lane & 16) ? r + 8 : r);
                int col = n0 + j * 16 + (lane & 15);
                C[(size_t)row * N + col] = acc[i][j][r];
            }
}

// ---------------------------------------------------------------------------
// RoPE on Q,K; convert to bf16 in (B,H,L,D) layout
// ---------------------------------------------------------------------------
__global__ void rope_qk(const float* __restrict__ qf, const float* __restrict__ kf,
                        const float* __restrict__ cosT, const float* __restrict__ sinT,
                        bf16_t* __restrict__ qb, bf16_t* __restrict__ kb) {
    int idx = blockIdx.x * blockDim.x + threadIdx.x;   // linear over (B,L,H,D)
    if (idx >= Bc * Lc * HIDc) return;
    int d = idx & (Dc - 1);
    int h = (idx >> 6) & (Hc - 1);
    int l = (idx >> 10) & (Lc - 1);
    int b = idx >> 21;
    float c = cosT[l * Dc + d];
    float s = sinT[l * Dc + d];
    float q = qf[idx], k = kf[idx];
    float qo, ko;
    if (d < Dc / 2) { qo = q * c - qf[idx + Dc / 2] * s; ko = k * c - kf[idx + Dc / 2] * s; }
    else            { qo = q * c + qf[idx - Dc / 2] * s; ko = k * c + kf[idx - Dc / 2] * s; }
    size_t dst = (((size_t)b * Hc + h) * Lc + l) * Dc + d;
    qb[dst] = (bf16_t)qo;
    kb[dst] = (bf16_t)ko;
}

// V: f32 (B,L,H,D) -> bf16 (B,H,D,L)  (d-major so P*V B-fragments are b128)
__global__ void v_to_dmajor(const float* __restrict__ vf, bf16_t* __restrict__ vt) {
    int idx = blockIdx.x * blockDim.x + threadIdx.x;
    if (idx >= Bc * Lc * HIDc) return;
    int d = idx & (Dc - 1);
    int h = (idx >> 6) & (Hc - 1);
    int l = (idx >> 10) & (Lc - 1);
    int b = idx >> 21;
    vt[(((size_t)b * Hc + h) * Dc + d) * Lc + l] = (bf16_t)vf[idx];
}

// ---------------------------------------------------------------------------
// Causal flash attention. One wave per 16-row Q tile; 4 waves/block.
// Q,K: (B,H,L,D) bf16;  V: (B,H,D,L) bf16;  O: (B,L,H,D) f32
// grid = (L/64, B*H), block = 128
// ---------------------------------------------------------------------------
__global__ __launch_bounds__(128)
void attn_fwd(const bf16_t* __restrict__ Q, const bf16_t* __restrict__ Kt,
              const bf16_t* __restrict__ Vt, float* __restrict__ O) {
    __shared__ bf16_t psh[4][16 * 32];   // per-wave P staging (C-layout -> A-layout)
    const int lane = threadIdx.x & 31;
    const int wave = threadIdx.x >> 5;
    bf16_t* pb = psh[wave];

    const int bh = blockIdx.y;
    const int bq = bh / Hc, hq = bh % Hc;
    const int q0 = blockIdx.x * 64 + wave * 16;

    const bf16_t* Qb = Q  + (size_t)bh * Lc * Dc;
    const bf16_t* Kb = Kt + (size_t)bh * Lc * Dc;
    const bf16_t* Vb = Vt + (size_t)bh * Dc * Lc;

    Frag qa0 = load_frag(Qb, Dc, q0 + (lane & 15), 0,  lane);
    Frag qa1 = load_frag(Qb, Dc, q0 + (lane & 15), 32, lane);

    const v8f vz = {0.f,0.f,0.f,0.f,0.f,0.f,0.f,0.f};
    v8f o[4] = {vz, vz, vz, vz};
    float mr[8], lr[8];
#pragma unroll
    for (int r = 0; r < 8; ++r) { mr[r] = -__builtin_inff(); lr[r] = 0.f; }

    const int qhi = q0 + 15;
    for (int k0 = 0; k0 <= qhi; k0 += 32) {
        // S = Q * K^T  (two 16-col tiles; WMMA K-dim = d, 2 chunks of 32)
        v8f s0 = vz, s1 = vz;
        Frag kfr;
        kfr = load_frag(Kb, Dc, k0 + (lane & 15), 0,  lane);  s0 = wmma_bf16(qa0, kfr, s0);
        kfr = load_frag(Kb, Dc, k0 + (lane & 15), 32, lane);  s0 = wmma_bf16(qa1, kfr, s0);
        kfr = load_frag(Kb, Dc, k0 + 16 + (lane & 15), 0,  lane); s1 = wmma_bf16(qa0, kfr, s1);
        kfr = load_frag(Kb, Dc, k0 + 16 + (lane & 15), 32, lane); s1 = wmma_bf16(qa1, kfr, s1);

        const int n0c = k0 + (lane & 15);
        const int n1c = n0c + 16;
#pragma unroll
        for (int r = 0; r < 8; ++r) {
            const int rowl = (lane & 16) ? r + 8 : r;
            const int rowg = q0 + rowl;
            float e0 = s0[r] * 0.125f; if (n0c > rowg) e0 = -__builtin_inff();
            float e1 = s1[r] * 0.125f; if (n1c > rowg) e1 = -__builtin_inff();
            // row max across the 16 lanes of this half (xor masks <16 stay in half)
            float mx = fmaxf(e0, e1);
            mx = fmaxf(mx, __shfl_xor(mx, 1));
            mx = fmaxf(mx, __shfl_xor(mx, 2));
            mx = fmaxf(mx, __shfl_xor(mx, 4));
            mx = fmaxf(mx, __shfl_xor(mx, 8));
            float mnew = fmaxf(mr[r], mx);
            float p0 = __expf(e0 - mnew);
            float p1 = __expf(e1 - mnew);
            float sum = p0 + p1;
            sum += __shfl_xor(sum, 1);
            sum += __shfl_xor(sum, 2);
            sum += __shfl_xor(sum, 4);
            sum += __shfl_xor(sum, 8);
            float alpha = __expf(mr[r] - mnew);
            lr[r] = lr[r] * alpha + sum;
            mr[r] = mnew;
            o[0][r] = o[0][r] * alpha;
            o[1][r] = o[1][r] * alpha;
            o[2][r] = o[2][r] * alpha;
            o[3][r] = o[3][r] * alpha;
            pb[rowl * 32 +      (lane & 15)] = (bf16_t)p0;
            pb[rowl * 32 + 16 + (lane & 15)] = (bf16_t)p1;
        }
        // DS ops within a wave are in-order; wait own stores + stop reordering.
        asm volatile("s_wait_dscnt 0x0" ::: "memory");
        Frag pf = load_frag(pb, 32, lane & 15, 0, lane);
        // O += P * V  (WMMA K-dim = 32 k-positions; 4 tiles of 16 d-columns)
#pragma unroll
        for (int j = 0; j < 4; ++j) {
            Frag vfr = load_frag(Vb, Lc, j * 16 + (lane & 15), k0, lane);
            o[j] = wmma_bf16(pf, vfr, o[j]);
        }
        asm volatile("" ::: "memory");
    }

    // normalize and write (B,L,H,D)
#pragma unroll
    for (int r = 0; r < 8; ++r) {
        const int rowl = (lane & 16) ? r + 8 : r;
        const int rowg = q0 + rowl;
        const float inv = 1.0f / lr[r];
        float* orow = O + (((size_t)bq * Lc + rowg) * Hc + hq) * Dc;
#pragma unroll
        for (int j = 0; j < 4; ++j)
            orow[j * 16 + (lane & 15)] = o[j][r] * inv;
    }
}

// ---------------------------------------------------------------------------
// Host launch
// ---------------------------------------------------------------------------
extern "C" void kernel_launch(void* const* d_in, const int* in_sizes, int n_in,
                              void* d_out, int out_size, void* d_ws, size_t ws_size,
                              hipStream_t stream) {
    (void)in_sizes; (void)n_in; (void)out_size; (void)ws_size;
    const float* x    = (const float*)d_in[0];
    const float* Wq   = (const float*)d_in[1];
    const float* Wk   = (const float*)d_in[2];
    const float* Wv   = (const float*)d_in[3];
    const float* Wo   = (const float*)d_in[4];
    const float* cosT = (const float*)d_in[5];
    const float* sinT = (const float*)d_in[6];
    float* out = (float*)d_out;

    char* ws = (char*)d_ws;
    const size_t MB = 1u << 20;
    bf16_t* xb    = (bf16_t*)(ws + 0 * MB);     // 8 MB
    bf16_t* wqb   = (bf16_t*)(ws + 8 * MB);     // 2 MB
    bf16_t* wkb   = (bf16_t*)(ws + 10 * MB);
    bf16_t* wvb   = (bf16_t*)(ws + 12 * MB);
    bf16_t* wob   = (bf16_t*)(ws + 14 * MB);
    float*  qf    = (float*) (ws + 16 * MB);    // 16 MB
    float*  kf    = (float*) (ws + 32 * MB);
    float*  vf    = (float*) (ws + 48 * MB);
    bf16_t* qb    = (bf16_t*)(ws + 64 * MB);    // 8 MB
    bf16_t* kb    = (bf16_t*)(ws + 72 * MB);
    bf16_t* vt    = (bf16_t*)(ws + 80 * MB);
    float*  attnf = (float*) (ws + 88 * MB);    // 16 MB
    bf16_t* ab    = (bf16_t*)(ws + 104 * MB);   // 8 MB  (total 112 MB)

    const int nX = Mc * HIDc;       // 4,194,304
    const int nW = HIDc * HIDc;     // 1,048,576

    cvt_f32_bf16<<<nX / 256, 256, 0, stream>>>(x,  xb,  nX);
    cvt_f32_bf16<<<nW / 256, 256, 0, stream>>>(Wq, wqb, nW);
    cvt_f32_bf16<<<nW / 256, 256, 0, stream>>>(Wk, wkb, nW);
    cvt_f32_bf16<<<nW / 256, 256, 0, stream>>>(Wv, wvb, nW);
    cvt_f32_bf16<<<nW / 256, 256, 0, stream>>>(Wo, wob, nW);

    dim3 ggrid(HIDc / 128, Mc / 64);            // (8, 64)
    gemm_nt_wmma<<<ggrid, 128, 0, stream>>>(xb, wqb, qf);
    gemm_nt_wmma<<<ggrid, 128, 0, stream>>>(xb, wkb, kf);
    gemm_nt_wmma<<<ggrid, 128, 0, stream>>>(xb, wvb, vf);

    rope_qk<<<nX / 256, 256, 0, stream>>>(qf, kf, cosT, sinT, qb, kb);
    v_to_dmajor<<<nX / 256, 256, 0, stream>>>(vf, vt);

    dim3 agrid(Lc / 64, Bc * Hc);               // (32, 32)
    attn_fwd<<<agrid, 128, 0, stream>>>(qb, kb, vt, attnf);

    cvt_f32_bf16<<<nX / 256, 256, 0, stream>>>(attnf, ab, nX);
    gemm_nt_wmma<<<ggrid, 128, 0, stream>>>(ab, wob, out);
}
